// MultiHeadAttention_22290880266778
// MI455X (gfx1250) — compile-verified, matrix operand order checked by reading, not run on hardware
//
#include <hip/hip_runtime.h>

// Problem sizes (compile-time constants from the reference).
#define B_  4
#define N_  2048
#define D_  1024
#define H_  16
#define DH_ 64
#define M_  (B_ * N_)   // 8192 rows of x
#define NE_ (3 * D_)    // 3072 fused QKV output features

typedef __attribute__((ext_vector_type(16))) __bf16       v16bf;
typedef __attribute__((ext_vector_type(8)))  float        v8f;
typedef __attribute__((ext_vector_type(4)))  unsigned int v4u;

// 32-byte A/B fragment for v_wmma_f32_16x16x32_bf16 (16 bf16 per lane).
union Frag { v4u u[2]; v16bf h; };

__device__ __forceinline__ unsigned short f32_bf16(float f) {
  unsigned int u = __float_as_uint(f);
  u += 0x7FFFu + ((u >> 16) & 1u);   // round-to-nearest-even
  return (unsigned short)(u >> 16);
}

__device__ __forceinline__ v8f wmma_bf16(const Frag& a, const Frag& b, v8f c) {
  // (neg_a, A, neg_b, B, c_mod, C, reuse_a, reuse_b)
  return __builtin_amdgcn_wmma_f32_16x16x32_bf16(false, a.h, false, b.h,
                                                 (short)0, c, false, false);
}

// gfx1250 async DMA: 16B global -> LDS, tracked by ASYNCcnt.
// vdst VGPR carries the LDS byte address (low 32 bits of the flat shared
// pointer == allocation-relative LDS offset per the aperture mapping).
__device__ __forceinline__ void async_copy16(void* lds_ptr,
                                             const unsigned short* g) {
  unsigned lds_off = (unsigned)(unsigned long long)lds_ptr;
  asm volatile("global_load_async_to_lds_b128 %0, %1, off"
               : : "v"(lds_off), "v"(g) : "memory");
}

__device__ __forceinline__ void wait_async0() {
#if __has_builtin(__builtin_amdgcn_s_wait_asynccnt)
  __builtin_amdgcn_s_wait_asynccnt(0);
#else
  asm volatile("s_wait_asynccnt 0x0" ::: "memory");
#endif
}

// ---------------------------------------------------------------- convert ---
__global__ void cvt_kernel(const float* __restrict__ src,
                           unsigned short* __restrict__ dst, int n) {
  int i = blockIdx.x * blockDim.x + threadIdx.x;
  if (i < n) dst[i] = f32_bf16(src[i]);
}

// ------------------------------------------------------------- GEMM core ----
// C[64x32] wave tile of A[M,1024] * W[Nc,1024]^T, bf16 in, f32 accum.
__device__ __forceinline__ void gemm_core(const unsigned short* __restrict__ A,
                                          const unsigned short* __restrict__ W,
                                          v8f (&acc)[4][2], int lane) {
  const int lrow = lane & 15;
  const int kh   = (lane >> 4) << 3;   // 0 or 8
#pragma unroll
  for (int mt = 0; mt < 4; ++mt)
#pragma unroll
    for (int nt = 0; nt < 2; ++nt) {
      v8f z = {0.f, 0.f, 0.f, 0.f, 0.f, 0.f, 0.f, 0.f};
      acc[mt][nt] = z;
    }
  for (int k = 0; k < D_; k += 32) {
    const unsigned short* w0 = W + (size_t)(lrow)      * D_ + k + kh;
    const unsigned short* w1 = W + (size_t)(16 + lrow) * D_ + k + kh;
    Frag b0, b1;
    b0.u[0] = *(const v4u*)(w0);      b0.u[1] = *(const v4u*)(w0 + 16);
    b1.u[0] = *(const v4u*)(w1);      b1.u[1] = *(const v4u*)(w1 + 16);
    if (k + 32 < D_) {                 // near-scope prefetch of next K step
      __builtin_prefetch(w0 + 32, 0, 3);
      __builtin_prefetch(w1 + 32, 0, 3);
    }
#pragma unroll
    for (int mt = 0; mt < 4; ++mt) {
      const unsigned short* a = A + (size_t)(mt * 16 + lrow) * D_ + k + kh;
      Frag af;
      af.u[0] = *(const v4u*)(a);     af.u[1] = *(const v4u*)(a + 16);
      acc[mt][0] = wmma_bf16(af, b0, acc[mt][0]);
      acc[mt][1] = wmma_bf16(af, b1, acc[mt][1]);
    }
  }
}

// ------------------------------------------------------ fused QKV GEMM ------
// Wb = [3072,1024] bf16 (Wq;Wk;Wv rows). Writes Q,K row-major [8192,1024],
// V transposed per head: Vt[((b*H+h)*DH+dh)*N + n].
__global__ __launch_bounds__(256) void qkv_kernel(
    const unsigned short* __restrict__ Xb,
    const unsigned short* __restrict__ Wb,
    unsigned short* __restrict__ Qo,
    unsigned short* __restrict__ Ko,
    unsigned short* __restrict__ Vt) {
  const int wid   = threadIdx.x >> 5;
  const int lane  = threadIdx.x & 31;
  const int mbase = blockIdx.x * 128 + (wid >> 2) * 64;   // 2 waves along M
  const int nbase = blockIdx.y * 128 + (wid & 3) * 32;    // 4 waves along N
  v8f acc[4][2];
  gemm_core(Xb + (size_t)mbase * D_, Wb + (size_t)nbase * D_, acc, lane);
  const int lrow = lane & 15;
  const int kh   = (lane >> 4) << 3;
#pragma unroll
  for (int mt = 0; mt < 4; ++mt)
#pragma unroll
    for (int nt = 0; nt < 2; ++nt)
#pragma unroll
      for (int r = 0; r < 8; ++r) {
        int m = mbase + mt * 16 + r + kh;    // D-frag row = r + 8*(lane>=16)
        int e = nbase + nt * 16 + lrow;      // D-frag col = lane&15
        unsigned short v = f32_bf16(acc[mt][nt][r]);
        if (e < D_) {
          Qo[(size_t)m * D_ + e] = v;
        } else if (e < 2 * D_) {
          Ko[(size_t)m * D_ + (e - D_)] = v;
        } else {
          int e2 = e - 2 * D_;
          int hh = e2 >> 6, dh = e2 & 63;
          int bb = m >> 11, n = m & (N_ - 1);
          Vt[((size_t)((bb * H_ + hh) * DH_ + dh)) * N_ + n] = v;
        }
      }
}

// ------------------------------------------------- flash-style attention ----
// Block = (b,h,64-query rows); 4 waves x 16 rows; key blocks of 64.
// K/V tiles stream into double-buffered LDS via global_load_async_to_lds.
__global__ __launch_bounds__(128) void attn_kernel(
    const unsigned short* __restrict__ Q,
    const unsigned short* __restrict__ Km,
    const unsigned short* __restrict__ Vt,
    unsigned short* __restrict__ AO) {
  const int qblk = blockIdx.x;
  const int h    = blockIdx.y;
  const int b    = blockIdx.z;
  const int tid  = threadIdx.x;
  const int wid  = tid >> 5;
  const int lane = tid & 31;
  const int lrow = lane & 15;
  const int kh   = (lane >> 4) << 3;   // 0 or 8

  // Padded rows (72 halfs = 36 dwords): 16 consecutive rows hit distinct banks.
  __shared__ __align__(16) unsigned short Kt[2][64][72];
  __shared__ __align__(16) unsigned short Vs[2][64][72];
  __shared__ __align__(16) unsigned short Plds[4][16][72];

  const unsigned short* Kbase = Km + (size_t)(b * N_) * D_ + h * DH_;
  const unsigned short* Vbase = Vt + (size_t)((b * H_ + h) * DH_) * N_;

  // Cooperative async stage of one 64-key block (8KB K + 8KB V).
  auto issue_tiles = [&](int kb, int buf) {
#pragma unroll
    for (int i = 0; i < 4; ++i) {
      int c   = tid + i * 128;        // 0..511 chunk id
      int row = c >> 3;               // 0..63
      int cc  = (c & 7) * 8;          // 0,8,..,56 (halfs)
      async_copy16(&Kt[buf][row][cc], Kbase + (size_t)(kb + row) * D_ + cc);
      async_copy16(&Vs[buf][row][cc], Vbase + (size_t)row * N_ + kb + cc);
    }
  };

  issue_tiles(0, 0);

  const int q0 = qblk * 64 + wid * 16;
  Frag qa0, qa1;   // 16 query rows x DH=64 as two K=32 A-fragments
  {
    const unsigned short* qb = Q + ((size_t)(b * N_ + q0 + lrow)) * D_ + h * DH_;
    qa0.u[0] = *(const v4u*)(qb + kh);
    qa0.u[1] = *(const v4u*)(qb + 16 + kh);
    qa1.u[0] = *(const v4u*)(qb + 32 + kh);
    qa1.u[1] = *(const v4u*)(qb + 48 + kh);
  }

  v8f acc[4];      // 16 rows x 64 dh output, 4 n-tiles
#pragma unroll
  for (int t = 0; t < 4; ++t) { v8f z = {0,0,0,0,0,0,0,0}; acc[t] = z; }
  float rmax[8], rsum[8];
#pragma unroll
  for (int r = 0; r < 8; ++r) { rmax[r] = -3.0e38f; rsum[r] = 0.f; }

  const float sc = 0.125f * 1.44269504088896f;   // (1/sqrt(DH)) * log2(e)

  for (int it = 0; it < N_ / 64; ++it) {
    const int buf = it & 1;
    wait_async0();        // our tile DMAs for `buf` have landed
    __syncthreads();      // ...for every wave; also closes last iter's reads
    if (it + 1 < N_ / 64) issue_tiles((it + 1) * 64, buf ^ 1);

    // ---- S = Q K^T for a 16x64 score tile (K fragments from LDS)
    v8f s[4];
#pragma unroll
    for (int nt = 0; nt < 4; ++nt) {
      const unsigned short* kp = &Kt[buf][nt * 16 + lrow][0];
      Frag k0, k1;
      k0.u[0] = *(const v4u*)(kp + kh);
      k0.u[1] = *(const v4u*)(kp + 16 + kh);
      k1.u[0] = *(const v4u*)(kp + 32 + kh);
      k1.u[1] = *(const v4u*)(kp + 48 + kh);
      v8f z = {0,0,0,0,0,0,0,0};
      s[nt] = wmma_bf16(qa0, k0, z);
      s[nt] = wmma_bf16(qa1, k1, s[nt]);
    }
    // ---- online softmax (log2 domain); rows live across 16-lane groups
    float mnew[8];
#pragma unroll
    for (int r = 0; r < 8; ++r) {
      float m = rmax[r];
#pragma unroll
      for (int nt = 0; nt < 4; ++nt) m = fmaxf(m, s[nt][r] * sc);
      mnew[r] = m;
    }
#pragma unroll
    for (int mk = 1; mk <= 8; mk <<= 1)
#pragma unroll
      for (int r = 0; r < 8; ++r)
        mnew[r] = fmaxf(mnew[r], __shfl_xor(mnew[r], mk, 32));
#pragma unroll
    for (int r = 0; r < 8; ++r) {
      float corr = __builtin_exp2f(rmax[r] - mnew[r]);
      rmax[r] = mnew[r];
      rsum[r] *= corr;
#pragma unroll
      for (int t = 0; t < 4; ++t) acc[t][r] *= corr;
    }
    // ---- P = exp2(S*sc - m): D-layout -> LDS -> A-layout
#pragma unroll
    for (int nt = 0; nt < 4; ++nt)
#pragma unroll
      for (int r = 0; r < 8; ++r) {
        float p = __builtin_exp2f(s[nt][r] * sc - rmax[r]);
        rsum[r] += p;                       // lane-partial row sum
        Plds[wid][r + kh][nt * 16 + lrow] = f32_bf16(p);
      }
    __syncthreads();
    Frag pa0, pa1;
    {
      const unsigned short* pl = &Plds[wid][lrow][0];
      pa0.u[0] = *(const v4u*)(pl + kh);
      pa0.u[1] = *(const v4u*)(pl + 16 + kh);
      pa1.u[0] = *(const v4u*)(pl + 32 + kh);
      pa1.u[1] = *(const v4u*)(pl + 48 + kh);
    }
    // ---- O += P * V (V fragments from LDS; V stored transposed)
#pragma unroll
    for (int t = 0; t < 4; ++t) {
      const unsigned short* vb = &Vs[buf][t * 16 + lrow][0];
      Frag v0, v1;
      v0.u[0] = *(const v4u*)(vb + kh);
      v0.u[1] = *(const v4u*)(vb + 16 + kh);
      v1.u[0] = *(const v4u*)(vb + 32 + kh);
      v1.u[1] = *(const v4u*)(vb + 48 + kh);
      acc[t] = wmma_bf16(pa0, v0, acc[t]);
      acc[t] = wmma_bf16(pa1, v1, acc[t]);
    }
  }
  // ---- finalize: reduce row sums across the 16 lanes, normalize, store bf16
#pragma unroll
  for (int mk = 1; mk <= 8; mk <<= 1)
#pragma unroll
    for (int r = 0; r < 8; ++r) rsum[r] += __shfl_xor(rsum[r], mk, 32);
#pragma unroll
  for (int r = 0; r < 8; ++r) {
    float inv = 1.0f / rsum[r];
    int m = b * N_ + q0 + r + kh;
#pragma unroll
    for (int t = 0; t < 4; ++t) {
      int e = h * DH_ + t * 16 + lrow;
      AO[(size_t)m * D_ + e] = f32_bf16(acc[t][r] * inv);
    }
  }
}

// --------------------------------------------------- output projection ------
__global__ __launch_bounds__(256) void oproj_kernel(
    const unsigned short* __restrict__ Ab,
    const unsigned short* __restrict__ Wob,
    float* __restrict__ Out) {
  const int wid   = threadIdx.x >> 5;
  const int lane  = threadIdx.x & 31;
  const int mbase = blockIdx.x * 128 + (wid >> 2) * 64;
  const int nbase = blockIdx.y * 128 + (wid & 3) * 32;
  v8f acc[4][2];
  gemm_core(Ab + (size_t)mbase * D_, Wob + (size_t)nbase * D_, acc, lane);
  const int lrow = lane & 15;
  const int kh   = (lane >> 4) << 3;
#pragma unroll
  for (int mt = 0; mt < 4; ++mt)
#pragma unroll
    for (int nt = 0; nt < 2; ++nt)
#pragma unroll
      for (int r = 0; r < 8; ++r)
        Out[(size_t)(mbase + mt * 16 + r + kh) * D_ + (nbase + nt * 16 + lrow)] =
            acc[mt][nt][r];
}

// ------------------------------------------------------------- launcher -----
extern "C" void kernel_launch(void* const* d_in, const int* in_sizes, int n_in,
                              void* d_out, int out_size, void* d_ws, size_t ws_size,
                              hipStream_t stream) {
  (void)in_sizes; (void)n_in; (void)out_size; (void)ws_size;
  const float* x  = (const float*)d_in[0];
  // d_in[1] = mask (dead code in reference)
  const float* Wq = (const float*)d_in[2];
  const float* Wk = (const float*)d_in[3];
  const float* Wv = (const float*)d_in[4];
  const float* Wo = (const float*)d_in[5];

  char* ws = (char*)d_ws;                       // bytes       cumulative
  unsigned short* Xb  = (unsigned short*)(ws);             //  0   .. 16 MiB
  unsigned short* Wb  = (unsigned short*)(ws + 16777216);  // QKV weights 6 MiB
  unsigned short* Wob = (unsigned short*)(ws + 23068672);  // Wo 2 MiB
  unsigned short* Qo  = (unsigned short*)(ws + 25165824);  // 16 MiB
  unsigned short* Ko  = (unsigned short*)(ws + 41943040);  // 16 MiB
  unsigned short* Vt  = (unsigned short*)(ws + 58720256);  // 16 MiB
  unsigned short* AO  = (unsigned short*)(ws + 75497472);  // 16 MiB

  const int nx = M_ * D_;     // 8388608
  const int nw = D_ * D_;     // 1048576
  cvt_kernel<<<(nx + 255) / 256, 256, 0, stream>>>(x,  Xb,          nx);
  cvt_kernel<<<(nw + 255) / 256, 256, 0, stream>>>(Wq, Wb,          nw);
  cvt_kernel<<<(nw + 255) / 256, 256, 0, stream>>>(Wk, Wb + nw,     nw);
  cvt_kernel<<<(nw + 255) / 256, 256, 0, stream>>>(Wv, Wb + 2 * nw, nw);
  cvt_kernel<<<(nw + 255) / 256, 256, 0, stream>>>(Wo, Wob,         nw);

  qkv_kernel<<<dim3(M_ / 128, NE_ / 128), 256, 0, stream>>>(Xb, Wb, Qo, Ko, Vt);
  attn_kernel<<<dim3(N_ / 64, H_, B_), 128, 0, stream>>>(Qo, Ko, Vt, AO);
  oproj_kernel<<<dim3(M_ / 128, D_ / 128), 256, 0, stream>>>(AO, Wob,
                                                             (float*)d_out);
}
